// VectorQuantiser_43696997269852
// MI455X (gfx1250) — matrix operand order repaired and checked by reading, not compiled, not run on hardware
//
#include <hip/hip_runtime.h>
#include <hip/hip_bf16.h>

// ---------------------------------------------------------------------------
// VQ codebook search on gfx1250 (CDNA5, wave32, WMMA + async-to-LDS)
//   z:            (8, 64, 4096) f32   -> 32768 rows x 64 features
//   embed_weight: (8192, 64)    f32
// Outputs (flat concat): z_q (2097152) | loss (1) | indices-as-f32 (32768) | perplexity (1)
// ---------------------------------------------------------------------------

#define NUM_EMBED   8192
#define EMBED_DIM   64
#define NROWS       32768          // 8 * 4096
#define H_DIM       4096
#define ZQ_ELEMS    2097152        // 8*64*4096
#define NTILES      (NUM_EMBED / 16)   // 512 code tiles of 16
#define CHUNK_TILES 8                  // tiles staged in LDS per step
#define NCHUNKS     (NTILES / CHUNK_TILES)  // 64
#define CHUNK_U16   (CHUNK_TILES * 2 * 512)   // 8192 u16 = 16 KB per chunk

typedef __attribute__((ext_vector_type(16))) __bf16 v16bf;
typedef __attribute__((ext_vector_type(8)))  float  v8f;

// ws layout (bytes)
#define WS_NEGNORM   0              // f32[8192]
#define WS_PACKEDB   32768          // u16[8192*64]  (WMMA B fragments, bf16 bits)
#define WS_IDX       1081344       // i32[32768]
#define WS_COUNTS    1212416       // f32[8192]
#define WS_PARTIAL   1245184       // f32[128]

static __device__ __forceinline__ unsigned short f2bf(float f) {
    unsigned int u = __float_as_uint(f);
    u += 0x7FFFu + ((u >> 16) & 1u);           // round-to-nearest-even
    return (unsigned short)(u >> 16);
}

// low 32 bits of the generic shared-aperture address == LDS byte offset
static __device__ __forceinline__ unsigned lds_off(const void* p) {
    return (unsigned)(unsigned long long)p;
}

static __device__ __forceinline__ void async_to_lds_b128(unsigned lds, unsigned long long gaddr) {
    asm volatile("global_load_async_to_lds_b128 %0, %1, off"
                 :: "v"(lds), "v"(gaddr) : "memory");
}
static __device__ __forceinline__ void async_to_lds_b32(unsigned lds, unsigned long long gaddr) {
    asm volatile("global_load_async_to_lds_b32 %0, %1, off"
                 :: "v"(lds), "v"(gaddr) : "memory");
}
static __device__ __forceinline__ void wait_asynccnt0() {
    asm volatile("s_wait_asynccnt 0x0" ::: "memory");
}

// ---------------- kernel 0: zero histogram + partials ----------------------
__global__ void vq_init(float* counts, float* partial) {
    int t = blockIdx.x * blockDim.x + threadIdx.x;
    if (t < NUM_EMBED) counts[t] = 0.0f;
    if (t < 128)       partial[t] = 0.0f;
}

// ---------------- kernel 1: -|e|^2 per code --------------------------------
__global__ void vq_negnorm(const float* __restrict__ embed, float* __restrict__ negnorm) {
    int m = blockIdx.x * blockDim.x + threadIdx.x;
    if (m >= NUM_EMBED) return;
    float s = 0.0f;
    #pragma unroll 8
    for (int k = 0; k < EMBED_DIM; ++k) {
        float e = embed[m * EMBED_DIM + k];
        s += e * e;
    }
    negnorm[m] = -s;
}

// ---------------- kernel 2: pack codebook into WMMA B fragments ------------
// B fragment (32x16 bf16): lane L, slot i -> K = i + (L<16?0:16), col = L&15.
// Fragment order: [(tile*2 + kfrag)*32 + lane]*16 + i
__global__ void vq_packB(const float* __restrict__ embed, unsigned short* __restrict__ packedB) {
    int t = blockIdx.x * blockDim.x + threadIdx.x;   // 512*2*32 = 32768 threads
    if (t >= NTILES * 2 * 32) return;
    int lane  = t & 31;
    int kf    = (t >> 5) & 1;
    int tile  = t >> 6;
    int code  = tile * 16 + (lane & 15);
    int kbase = kf * 32 + ((lane < 16) ? 0 : 16);
    unsigned short* dst = packedB + (size_t)t * 16;
    #pragma unroll
    for (int i = 0; i < 16; ++i)
        dst[i] = f2bf(embed[code * EMBED_DIM + kbase + i]);
}

// ---------------- kernel 3: WMMA GEMM + running argmax ---------------------
// grid 256 x 256 threads (8 waves); wave w handles row-block = wg*8 + w (16 rows).
// Double-buffered LDS filled by GLOBAL_LOAD_ASYNC_TO_LDS; one barrier per chunk.
__global__ __launch_bounds__(256)
void vq_argmax(const float* __restrict__ z,
               const unsigned short* __restrict__ packedB,
               const float* __restrict__ negnorm,
               int* __restrict__ idxOut) {
    __shared__ __align__(16) unsigned short sB[2][CHUNK_U16];   // 2 x 16 KB
    __shared__ __align__(16) float sNN[2][CHUNK_TILES * 16];    // 2 x 512 B

    const int tid  = threadIdx.x;
    const int lane = tid & 31;
    const int wid  = tid >> 5;
    const int block = blockIdx.x * 8 + wid;      // 0..2047
    const int row   = lane & 15;                 // A-matrix M for this lane
    const int n     = block * 16 + row;          // global query row
    const int b     = n >> 12;                   // n / 4096
    const int h     = n & (H_DIM - 1);

    // per-thread async-fill of one chunk into LDS buffer `buf`
    auto issue_chunk = [&](int chunk, int buf) {
        const char* src = (const char*)packedB + (size_t)chunk * (CHUNK_U16 * 2);
        unsigned lbase = lds_off(&sB[buf][0]);
        #pragma unroll
        for (int k = 0; k < 4; ++k) {
            int e = (tid + k * 256) * 16;        // byte offset, 16B granules
            async_to_lds_b128(lbase + e, (unsigned long long)(src + e));
        }
        if (tid < CHUNK_TILES * 16)
            async_to_lds_b32(lds_off(&sNN[buf][tid]),
                             (unsigned long long)(negnorm + chunk * (CHUNK_TILES * 16) + tid));
    };

    // ---- build A fragments (2*z in bf16, contiguous k order, same as packed B) ----
    // lane L slot i -> K = i + (L<16 ? 0 : 16)
    issue_chunk(0, 0);                           // overlap fill with A-gather
    v16bf a0, a1;
    const int koff = (lane < 16) ? 0 : 16;
    #pragma unroll
    for (int i = 0; i < 16; ++i) {
        int K = i + koff;
        float z0 = z[(size_t)b * (EMBED_DIM * H_DIM) + (size_t)K        * H_DIM + h];
        float z1 = z[(size_t)b * (EMBED_DIM * H_DIM) + (size_t)(K + 32) * H_DIM + h];
        a0[i] = (__bf16)(2.0f * z0);
        a1[i] = (__bf16)(2.0f * z1);
    }

    float best[8];
    int   bidx[8];
    #pragma unroll
    for (int r = 0; r < 8; ++r) { best[r] = -3.4e38f; bidx[r] = 0x7fffffff; }

    for (int chunk = 0; chunk < NCHUNKS; ++chunk) {
        const int buf = chunk & 1;
        wait_asynccnt0();                        // my async writes for this chunk done
        __syncthreads();                         // everyone's writes visible; prev reads done
        if (chunk + 1 < NCHUNKS)
            issue_chunk(chunk + 1, buf ^ 1);     // overlap next fill with compute

        #pragma unroll
        for (int tt = 0; tt < CHUNK_TILES; ++tt) {
            const v16bf b0 = *(const v16bf*)(&sB[buf][tt * 1024 +       lane * 16]);
            const v16bf b1 = *(const v16bf*)(&sB[buf][tt * 1024 + 512 + lane * 16]);
            // fold the -|e|^2 bias into the accumulator init (per-code == per-lane)
            const float nn = sNN[buf][tt * 16 + row];
            v8f c = {nn, nn, nn, nn, nn, nn, nn, nn};
            c = __builtin_amdgcn_wmma_f32_16x16x32_bf16(false, a0, false, b0,
                                                        (short)0, c, false, false);
            c = __builtin_amdgcn_wmma_f32_16x16x32_bf16(false, a1, false, b1,
                                                        (short)0, c, false, false);
            const int code = (chunk * CHUNK_TILES + tt) * 16 + row;
            // strict '>' keeps the first (lowest-index) max within this lane,
            // matching jnp.argmax; branch-free: 1 cmp + 2 cndmask per slot.
            #pragma unroll
            for (int r = 0; r < 8; ++r) {
                bool better = c[r] > best[r];
                best[r] = better ? c[r] : best[r];
                bidx[r] = better ? code : bidx[r];
            }
        }
    }

    // ---- argmax across the 16 lanes of each half-wave (columns) ----
    // tie-break (lowest index) applied here, once, outside the hot loop
    #pragma unroll
    for (int m = 1; m < 16; m <<= 1) {
        #pragma unroll
        for (int r = 0; r < 8; ++r) {
            float ov = __shfl_xor(best[r], m, 32);
            int   oi = __shfl_xor(bidx[r], m, 32);
            bool better = (ov > best[r]) || (ov == best[r] && oi < bidx[r]);
            best[r] = better ? ov : best[r];
            bidx[r] = better ? oi : bidx[r];
        }
    }
    if ((lane & 15) == 0) {
        int group = lane >> 4;                       // 0 -> rows 0..7, 1 -> rows 8..15
        #pragma unroll
        for (int r = 0; r < 8; ++r)
            idxOut[block * 16 + group * 8 + r] = bidx[r];
    }
}

// ---------------- kernel 4: gather z_q, per-block loss, histogram ----------
__global__ __launch_bounds__(256)
void vq_scatter(const float* __restrict__ z,
                const float* __restrict__ embed,
                const int* __restrict__ idxWs,
                float* __restrict__ out,
                float* __restrict__ counts,
                float* __restrict__ partial) {
    __shared__ float red[256];
    int n = blockIdx.x * blockDim.x + threadIdx.x;   // 0..32767
    int idx = idxWs[n];
    int b = n >> 12;
    int h = n & (H_DIM - 1);
    float ss = 0.0f;
    #pragma unroll 8
    for (int c = 0; c < EMBED_DIM; ++c) {
        float e  = embed[(size_t)idx * EMBED_DIM + c];
        size_t o = (size_t)b * (EMBED_DIM * H_DIM) + (size_t)c * H_DIM + h;
        float zv = z[o];
        out[o] = e;                                  // z_q_st == z_q numerically
        float d = e - zv;
        ss += d * d;
    }
    out[ZQ_ELEMS + 1 + n] = (float)idx;              // indices region (as f32)
    atomicAdd(&counts[idx], 1.0f);                   // exact: +1.0 counts

    red[threadIdx.x] = ss;
    __syncthreads();
    #pragma unroll
    for (int s = 128; s > 0; s >>= 1) {
        if (threadIdx.x < s) red[threadIdx.x] += red[threadIdx.x + s];
        __syncthreads();
    }
    if (threadIdx.x == 0) partial[blockIdx.x] = red[0];
}

// ---------------- kernel 5: finalize loss + perplexity ---------------------
__global__ void vq_finalize(const float* __restrict__ partial,
                            const float* __restrict__ counts,
                            float* __restrict__ out) {
    __shared__ float red[256];
    int t = threadIdx.x;

    // entropy: each thread sums a fixed strided set of bins (deterministic)
    float ent = 0.0f;
    for (int m = t; m < NUM_EMBED; m += 256) {
        float p = counts[m] * (1.0f / (float)NROWS);
        ent += p * __logf(p + 1e-10f);
    }
    red[t] = ent;
    __syncthreads();
    #pragma unroll
    for (int s = 128; s > 0; s >>= 1) {
        if (t < s) red[t] += red[t + s];
        __syncthreads();
    }
    if (t == 0) {
        float total = 0.0f;
        for (int i = 0; i < 128; ++i) total += partial[i];   // in-order, deterministic
        out[ZQ_ELEMS] = 1.25f * total / (float)ZQ_ELEMS;     // (1 + BETA) * mean(diff^2)
        out[ZQ_ELEMS + 1 + NROWS] = __expf(-red[0]);         // perplexity
    }
}

// ---------------------------------------------------------------------------
extern "C" void kernel_launch(void* const* d_in, const int* in_sizes, int n_in,
                              void* d_out, int out_size, void* d_ws, size_t ws_size,
                              hipStream_t stream) {
    const float* z     = (const float*)d_in[0];
    const float* embed = (const float*)d_in[1];
    float* out = (float*)d_out;
    char*  ws  = (char*)d_ws;

    float*          negnorm = (float*)(ws + WS_NEGNORM);
    unsigned short* packedB = (unsigned short*)(ws + WS_PACKEDB);
    int*            idxWs   = (int*)(ws + WS_IDX);
    float*          counts  = (float*)(ws + WS_COUNTS);
    float*          partial = (float*)(ws + WS_PARTIAL);

    vq_init    <<<33, 256, 0, stream>>>(counts, partial);
    vq_negnorm <<<32, 256, 0, stream>>>(embed, negnorm);
    vq_packB   <<<128, 256, 0, stream>>>(embed, packedB);
    vq_argmax  <<<256, 256, 0, stream>>>(z, packedB, negnorm, idxWs);
    vq_scatter <<<128, 256, 0, stream>>>(z, embed, idxWs, out, counts, partial);
    vq_finalize<<<1, 256, 0, stream>>>(partial, counts, out);
}
